// TriggerDetection_90744069030362
// MI455X (gfx1250) — matrix operand order repaired and checked by reading, not compiled, not run on hardware
//
#include <hip/hip_runtime.h>
#include <hip/hip_bf16.h>
#include <math.h>

// ---------------- problem constants ----------------
#define BB   32
#define LL   512
#define DD   1024
#define SS   256
#define EE   64
#define PPOS 17
#define COS_EPS 1e-8f

// ---------------- WMMA types ----------------
typedef __attribute__((ext_vector_type(16))) __bf16 v16bf;
typedef __attribute__((ext_vector_type(8)))  __bf16 v8bf;
typedef __attribute__((ext_vector_type(8)))  float  v8f;

#define WMMA_BF16(A, B, C) \
  __builtin_amdgcn_wmma_f32_16x16x32_bf16(false, (A), false, (B), (short)0, (C), false, false)

// Split f32 into bf16 hi + bf16 residual (x ~= hi + lo to ~fp32 precision).
__device__ inline void bf_split(float x, __bf16* ph, __bf16* pl) {
  __bf16 h = (__bf16)x;
  *ph = h;
  *pl = (__bf16)(x - (float)h);
}

// ---------------- fragment-major packed layouts ----------------
// All WMMA operands are stored pre-swizzled in fragment order (k = 16*hb + j) so that a wave's
// fragment load is two coalesced b128s at (tilebase + lane*32B):
//   tile layout: [tileIdx][ktile][lane(32)][16 bf16]
//
// A-operand element (row, k):  lane = (row&15) + 16*hb, pos = j
//   with kk=k&31, hb=kk>>4, j=kk&15                   (matches B packing k order)
__device__ __forceinline__ size_t a_pack_idx(int row, int k, int nkt) {
  int mt = row >> 4, l = row & 15, kt = k >> 5, kk = k & 31;
  int hb = (kk >> 4) & 1, j = kk & 15;
  return (((size_t)mt * nkt + kt) * 32 + l + 16 * hb) * 16 + j;
}
// B-operand element (n, k):  lane = (n&15) + 16*hb, pos = j
//   with kk=k&31, hb=kk>>4, j=kk&15                   (matches A packing k order)
__device__ __forceinline__ size_t b_pack_idx(int n, int k, int nkt) {
  int nt = n >> 4, l = n & 15, kt = k >> 5, kk = k & 31;
  int hb = (kk >> 4) & 1, j = kk & 15;
  return (((size_t)nt * nkt + kt) * 32 + l + 16 * hb) * 16 + j;
}

// Coalesced fragment load: lane reads its own contiguous 32B.
__device__ __forceinline__ v16bf load_frag(const __bf16* __restrict__ P,
                                           int tile, int kt, int nkt, int lane) {
  const __bf16* pr = P + (((size_t)tile * nkt + kt) * 32 + lane) * 16;
  v8bf c0 = *(const v8bf*)pr;
  v8bf c1 = *(const v8bf*)(pr + 8);
  return __builtin_shufflevector(c0, c1, 0, 1, 2, 3, 4, 5, 6, 7,
                                 8, 9, 10, 11, 12, 13, 14, 15);
}

// ---------------- stage kernels ----------------

// one block per (b,s): gather masked sent row -> f32 + A-packed hi/lo bf16
__global__ __launch_bounds__(256) void k_gather_sent(
    const float* __restrict__ emb, const int* __restrict__ idxs,
    float* __restrict__ sent, __bf16* __restrict__ sh, __bf16* __restrict__ sl) {
  int blk = blockIdx.x;            // global row = b*S + s
  int b = blk >> 8;
  int idx = idxs[blk];
  const float* src = emb + ((size_t)b * LL + (idx > 0 ? idx : 0)) * DD;
  float m = (idx > 0) ? 1.0f : 0.0f;
  size_t base = (size_t)blk * DD;
  for (int d = threadIdx.x; d < DD; d += blockDim.x) {
    float v = m * src[d];
    sent[base + d] = v;
    size_t pi = a_pack_idx(blk, d, DD / 32);
    bf_split(v, &sh[pi], &sl[pi]);
  }
}

// one block per (b,e): gather masked event row -> B-packed (for GEMM2) +
// B-packed transposed (for GEMM3) + ne
__global__ __launch_bounds__(256) void k_gather_event(
    const float* __restrict__ emb, const int* __restrict__ idxs,
    __bf16* __restrict__ evh, __bf16* __restrict__ evl,
    __bf16* __restrict__ evth, __bf16* __restrict__ evtl,
    float* __restrict__ ne) {
  int blk = blockIdx.x;            // b*E + e
  int b = blk >> 6, e = blk & 63;
  int idx = idxs[blk];
  const float* src = emb + ((size_t)b * LL + (idx > 0 ? idx : 0)) * DD;
  float m = (idx > 0) ? 1.0f : 0.0f;
  float ss = 0.0f;
  size_t base2 = (size_t)b * EE * DD;   // per-batch packed region (GEMM2 B)
  size_t base3 = (size_t)b * DD * EE;   // per-batch packed region (GEMM3 B)
  for (int d = threadIdx.x; d < DD; d += blockDim.x) {
    float v = m * src[d];
    __bf16 h, lo;
    bf_split(v, &h, &lo);
    size_t i2 = base2 + b_pack_idx(e, d, DD / 32);  // N=e, K=d
    evh[i2] = h;  evl[i2] = lo;
    size_t i3 = base3 + b_pack_idx(d, e, EE / 32);  // N=d, K=e
    evth[i3] = h; evtl[i3] = lo;
    ss += v * v;
  }
  __shared__ float red[256];
  red[threadIdx.x] = ss;
  __syncthreads();
  for (int off = 128; off > 0; off >>= 1) {
    if (threadIdx.x < (unsigned)off) red[threadIdx.x] += red[threadIdx.x + off];
    __syncthreads();
  }
  if (threadIdx.x == 0) ne[blk] = fmaxf(sqrtf(red[0]), COS_EPS);
}

// map_num[b] = 1 / (#valid event idxs)
__global__ void k_map_num(const int* __restrict__ idxs, float* __restrict__ mapn) {
  int b = threadIdx.x;
  if (b >= BB) return;
  int c = 0;
  for (int e = 0; e < EE; ++e) c += (idxs[b * EE + e] > 0) ? 1 : 0;
  mapn[b] = 1.0f / (float)(c > 0 ? c : 1);
}

// W -> B-packed Wt hi/lo. Iterate in PACKED order so writes are coalesced.
__global__ __launch_bounds__(256) void k_convert_w(
    const float* __restrict__ W, __bf16* __restrict__ wth, __bf16* __restrict__ wtl) {
  int i = blockIdx.x * 256 + threadIdx.x;   // packed index
  int j = i & 15;
  int lane = (i >> 4) & 31;
  int blk = i >> 9;                          // ntile*32 + kt
  int kt = blk & 31, ntile = blk >> 5;
  int n = ntile * 16 + (lane & 15);
  int k = kt * 32 + (lane >> 4) * 16 + j;
  bf_split(W[(size_t)k * DD + n], &wth[i], &wtl[i]);
}

// na[row] = max(||a[row,:]||, eps) from packed hi/lo ; one block per row
__global__ __launch_bounds__(256) void k_row_norm(
    const __bf16* __restrict__ ah, const __bf16* __restrict__ al,
    float* __restrict__ na) {
  int row = blockIdx.x;
  float ss = 0.0f;
  for (int d = threadIdx.x; d < DD; d += blockDim.x) {
    size_t pi = a_pack_idx(row, d, DD / 32);
    float v = (float)ah[pi] + (float)al[pi];
    ss += v * v;
  }
  __shared__ float red[256];
  red[threadIdx.x] = ss;
  __syncthreads();
  for (int off = 128; off > 0; off >>= 1) {
    if (threadIdx.x < (unsigned)off) red[threadIdx.x] += red[threadIdx.x + off];
    __syncthreads();
  }
  if (threadIdx.x == 0) na[row] = fmaxf(sqrtf(red[0]), COS_EPS);
}

// GEMM1: a[8192x1024] = sent @ W  (all operands fragment-packed, nkt=32)
// WG tile 128x128, 8 waves (4 M x 2 N), wave tile 32x64 (2x4 WMMA tiles).
__global__ __launch_bounds__(256) void k_gemm1(
    const __bf16* __restrict__ sh, const __bf16* __restrict__ sl,
    const __bf16* __restrict__ wth, const __bf16* __restrict__ wtl,
    __bf16* __restrict__ ah, __bf16* __restrict__ al) {
  const int NKT = DD / 32;
  int lane = threadIdx.x & 31;
  int wave = threadIdx.x >> 5;
  int wm = wave & 3, wn = wave >> 2;
  int row0 = blockIdx.y * 128 + wm * 32;
  int col0 = blockIdx.x * 128 + wn * 64;
  int mtb = row0 >> 4;   // mtiles mtb, mtb+1
  int ntb = col0 >> 4;   // ntiles ntb .. ntb+3

  v8f acc[2][4] = {};

  for (int kt = 0; kt < NKT; ++kt) {
    v16bf a0h = load_frag(sh, mtb,     kt, NKT, lane);
    v16bf a0l = load_frag(sl, mtb,     kt, NKT, lane);
    v16bf a1h = load_frag(sh, mtb + 1, kt, NKT, lane);
    v16bf a1l = load_frag(sl, mtb + 1, kt, NKT, lane);
    if (kt + 1 < NKT) {
      __builtin_prefetch(sh + (((size_t)mtb * NKT + kt + 1) * 32 + lane) * 16, 0, 3);
      __builtin_prefetch(wth + (((size_t)ntb * NKT + kt + 1) * 32 + lane) * 16, 0, 3);
    }
#pragma unroll
    for (int nt = 0; nt < 4; ++nt) {
      v16bf bh = load_frag(wth, ntb + nt, kt, NKT, lane);
      v16bf bl = load_frag(wtl, ntb + nt, kt, NKT, lane);
      acc[0][nt] = WMMA_BF16(a0h, bh, acc[0][nt]);
      acc[0][nt] = WMMA_BF16(a0l, bh, acc[0][nt]);
      acc[0][nt] = WMMA_BF16(a0h, bl, acc[0][nt]);
      acc[1][nt] = WMMA_BF16(a1h, bh, acc[1][nt]);
      acc[1][nt] = WMMA_BF16(a1l, bh, acc[1][nt]);
      acc[1][nt] = WMMA_BF16(a1h, bl, acc[1][nt]);
    }
  }

  // epilogue: write 'a' already A-packed for GEMM2 (K dim = col)
  int l = lane & 15, hb = (lane >> 4) & 1;
#pragma unroll
  for (int mt = 0; mt < 2; ++mt)
#pragma unroll
    for (int nt = 0; nt < 4; ++nt)
#pragma unroll
      for (int r = 0; r < 8; ++r) {
        int row = row0 + mt * 16 + hb * 8 + r;
        int col = col0 + nt * 16 + l;
        size_t o = a_pack_idx(row, col, DD / 32);
        bf_split(acc[mt][nt][r], &ah[o], &al[o]);
      }
}

// GEMM2 (per batch): logits = (a @ event^T) / (na*ne) -> A-packed hi/lo (K=e)
// WG tile 128x64, 8 waves (4 M x 2 N), wave tile 32x32 (2x2 WMMA tiles).
__global__ __launch_bounds__(256) void k_gemm2(
    const __bf16* __restrict__ ah, const __bf16* __restrict__ al,
    const __bf16* __restrict__ evh, const __bf16* __restrict__ evl,
    const float* __restrict__ na, const float* __restrict__ ne,
    __bf16* __restrict__ lgh, __bf16* __restrict__ lgl) {
  const int NKT = DD / 32;
  int b = blockIdx.y;
  const __bf16* Ah = ah + (size_t)b * SS * DD;
  const __bf16* Al = al + (size_t)b * SS * DD;
  const __bf16* Bh = evh + (size_t)b * EE * DD;
  const __bf16* Bl = evl + (size_t)b * EE * DD;
  int lane = threadIdx.x & 31;
  int wave = threadIdx.x >> 5;
  int wm = wave & 3, wn = wave >> 2;
  int row0 = blockIdx.x * 128 + wm * 32;
  int col0 = wn * 32;
  int mtb = row0 >> 4;
  int ntb = col0 >> 4;

  v8f acc[2][2] = {};

  for (int kt = 0; kt < NKT; ++kt) {
    v16bf a0h = load_frag(Ah, mtb,     kt, NKT, lane);
    v16bf a0l = load_frag(Al, mtb,     kt, NKT, lane);
    v16bf a1h = load_frag(Ah, mtb + 1, kt, NKT, lane);
    v16bf a1l = load_frag(Al, mtb + 1, kt, NKT, lane);
#pragma unroll
    for (int nt = 0; nt < 2; ++nt) {
      v16bf bh = load_frag(Bh, ntb + nt, kt, NKT, lane);
      v16bf bl = load_frag(Bl, ntb + nt, kt, NKT, lane);
      acc[0][nt] = WMMA_BF16(a0h, bh, acc[0][nt]);
      acc[0][nt] = WMMA_BF16(a0l, bh, acc[0][nt]);
      acc[0][nt] = WMMA_BF16(a0h, bl, acc[0][nt]);
      acc[1][nt] = WMMA_BF16(a1h, bh, acc[1][nt]);
      acc[1][nt] = WMMA_BF16(a1l, bh, acc[1][nt]);
      acc[1][nt] = WMMA_BF16(a1h, bl, acc[1][nt]);
    }
  }

  int l = lane & 15, hb = (lane >> 4) & 1;
#pragma unroll
  for (int mt = 0; mt < 2; ++mt)
#pragma unroll
    for (int nt = 0; nt < 2; ++nt)
#pragma unroll
      for (int r = 0; r < 8; ++r) {
        int row = row0 + mt * 16 + hb * 8 + r;
        int col = col0 + nt * 16 + l;
        float v = acc[mt][nt][r] / (na[b * SS + row] * ne[b * EE + col]);
        size_t o = (size_t)b * SS * EE + a_pack_idx(row, col, EE / 32);
        bf_split(v, &lgh[o], &lgl[o]);
      }
}

// GEMM3 (per batch): s2e = (logits @ event) * map_num[b]   (K = 64, nkt=2)
// WG tile 128x128, 8 waves (4 M x 2 N), wave tile 32x64 (2x4 WMMA tiles).
__global__ __launch_bounds__(256) void k_gemm3(
    const __bf16* __restrict__ lgh, const __bf16* __restrict__ lgl,
    const __bf16* __restrict__ evth, const __bf16* __restrict__ evtl,
    const float* __restrict__ mapn, float* __restrict__ s2e) {
  const int NKT = EE / 32;
  int b = blockIdx.z;
  const __bf16* Ah = lgh + (size_t)b * SS * EE;
  const __bf16* Al = lgl + (size_t)b * SS * EE;
  const __bf16* Bh = evth + (size_t)b * DD * EE;
  const __bf16* Bl = evtl + (size_t)b * DD * EE;
  int lane = threadIdx.x & 31;
  int wave = threadIdx.x >> 5;
  int wm = wave & 3, wn = wave >> 2;
  int row0 = blockIdx.y * 128 + wm * 32;
  int col0 = blockIdx.x * 128 + wn * 64;
  int mtb = row0 >> 4;
  int ntb = col0 >> 4;

  v8f acc[2][4] = {};

#pragma unroll
  for (int kt = 0; kt < NKT; ++kt) {
    v16bf a0h = load_frag(Ah, mtb,     kt, NKT, lane);
    v16bf a0l = load_frag(Al, mtb,     kt, NKT, lane);
    v16bf a1h = load_frag(Ah, mtb + 1, kt, NKT, lane);
    v16bf a1l = load_frag(Al, mtb + 1, kt, NKT, lane);
#pragma unroll
    for (int nt = 0; nt < 4; ++nt) {
      v16bf bh = load_frag(Bh, ntb + nt, kt, NKT, lane);
      v16bf bl = load_frag(Bl, ntb + nt, kt, NKT, lane);
      acc[0][nt] = WMMA_BF16(a0h, bh, acc[0][nt]);
      acc[0][nt] = WMMA_BF16(a0l, bh, acc[0][nt]);
      acc[0][nt] = WMMA_BF16(a0h, bl, acc[0][nt]);
      acc[1][nt] = WMMA_BF16(a1h, bh, acc[1][nt]);
      acc[1][nt] = WMMA_BF16(a1l, bh, acc[1][nt]);
      acc[1][nt] = WMMA_BF16(a1h, bl, acc[1][nt]);
    }
  }

  float scale = mapn[b];
  int l = lane & 15, hb = (lane >> 4) & 1;
#pragma unroll
  for (int mt = 0; mt < 2; ++mt)
#pragma unroll
    for (int nt = 0; nt < 4; ++nt)
#pragma unroll
      for (int r = 0; r < 8; ++r) {
        int row = row0 + mt * 16 + hb * 8 + r;
        int col = col0 + nt * 16 + l;
        s2e[((size_t)b * SS + row) * DD + col] = acc[mt][nt][r] * scale;
      }
}

// final: out[b,s,j] = sent . lw[j,0:1024] + s2e . lw[j,1024:2048] + lw[j,2048+pos] + lb[j]
__global__ __launch_bounds__(256) void k_final(
    const float* __restrict__ sent, const float* __restrict__ s2e,
    const float* __restrict__ lw, const float* __restrict__ lb,
    const int* __restrict__ pos_tag, float* __restrict__ out) {
  int blk = blockIdx.x;            // b*S + s
  const float* xs = sent + (size_t)blk * DD;
  const float* xa = s2e + (size_t)blk * DD;
  const int LWROW = 2 * DD + PPOS; // 2065
  float p0 = 0.0f, p1 = 0.0f;
  for (int d = threadIdx.x; d < DD; d += blockDim.x) {
    float vs = xs[d], va = xa[d];
    p0 += vs * lw[d]         + va * lw[DD + d];
    p1 += vs * lw[LWROW + d] + va * lw[LWROW + DD + d];
  }
  __shared__ float r0[256], r1[256];
  r0[threadIdx.x] = p0;
  r1[threadIdx.x] = p1;
  __syncthreads();
  for (int off = 128; off > 0; off >>= 1) {
    if (threadIdx.x < (unsigned)off) {
      r0[threadIdx.x] += r0[threadIdx.x + off];
      r1[threadIdx.x] += r1[threadIdx.x + off];
    }
    __syncthreads();
  }
  if (threadIdx.x == 0) {
    int p = pos_tag[blk];
    if (p < 0) p = PPOS - 1;
    out[(size_t)blk * 2 + 0] = r0[0] + lw[2 * DD + p] + lb[0];
    out[(size_t)blk * 2 + 1] = r1[0] + lw[LWROW + 2 * DD + p] + lb[1];
  }
}

// ---------------- workspace layout (byte offsets, 256B aligned) ----------------
static constexpr size_t alignB(size_t x) { return (x + 255) & ~(size_t)255; }
static constexpr size_t SZ_SENT  = (size_t)BB * SS * DD * 4;   // f32
static constexpr size_t SZ_S2E   = (size_t)BB * SS * DD * 4;   // f32
static constexpr size_t SZ_NA    = (size_t)BB * SS * 4;
static constexpr size_t SZ_NE    = (size_t)BB * EE * 4;
static constexpr size_t SZ_MAPN  = (size_t)BB * 4;
static constexpr size_t SZ_SD_BF = (size_t)BB * SS * DD * 2;   // bf16 packed [B*S, D]
static constexpr size_t SZ_W_BF  = (size_t)DD * DD * 2;        // bf16 packed [D, D]
static constexpr size_t SZ_ED_BF = (size_t)BB * EE * DD * 2;   // bf16 packed per batch
static constexpr size_t SZ_LG_BF = (size_t)BB * SS * EE * 2;   // bf16 packed per batch

static constexpr size_t O_SENT   = 0;
static constexpr size_t O_S2E    = alignB(O_SENT + SZ_SENT);
static constexpr size_t O_NA     = alignB(O_S2E + SZ_S2E);
static constexpr size_t O_NE     = alignB(O_NA + SZ_NA);
static constexpr size_t O_MAPN   = alignB(O_NE + SZ_NE);
static constexpr size_t O_SENT_H = alignB(O_MAPN + SZ_MAPN);
static constexpr size_t O_SENT_L = alignB(O_SENT_H + SZ_SD_BF);
static constexpr size_t O_A_H    = alignB(O_SENT_L + SZ_SD_BF);
static constexpr size_t O_A_L    = alignB(O_A_H + SZ_SD_BF);
static constexpr size_t O_WT_H   = alignB(O_A_L + SZ_SD_BF);
static constexpr size_t O_WT_L   = alignB(O_WT_H + SZ_W_BF);
static constexpr size_t O_EV_H   = alignB(O_WT_L + SZ_W_BF);
static constexpr size_t O_EV_L   = alignB(O_EV_H + SZ_ED_BF);
static constexpr size_t O_EVT_H  = alignB(O_EV_L + SZ_ED_BF);
static constexpr size_t O_EVT_L  = alignB(O_EVT_H + SZ_ED_BF);
static constexpr size_t O_LOG_H  = alignB(O_EVT_L + SZ_ED_BF);
static constexpr size_t O_LOG_L  = alignB(O_LOG_H + SZ_LG_BF);

extern "C" void kernel_launch(void* const* d_in, const int* in_sizes, int n_in,
                              void* d_out, int out_size, void* d_ws, size_t ws_size,
                              hipStream_t stream) {
  (void)in_sizes; (void)n_in; (void)out_size; (void)ws_size;

  const float* emb      = (const float*)d_in[0];
  const float* W        = (const float*)d_in[1];
  const float* lin_w    = (const float*)d_in[2];
  const float* lin_b    = (const float*)d_in[3];
  const int*   idx_sent = (const int*)d_in[4];
  const int*   idx_evt  = (const int*)d_in[5];
  const int*   pos_tag  = (const int*)d_in[6];
  float*       out      = (float*)d_out;

  char* ws = (char*)d_ws;
  float*  sent   = (float*)(ws + O_SENT);
  float*  s2e    = (float*)(ws + O_S2E);
  float*  na     = (float*)(ws + O_NA);
  float*  ne     = (float*)(ws + O_NE);
  float*  mapn   = (float*)(ws + O_MAPN);
  __bf16* sent_h = (__bf16*)(ws + O_SENT_H);
  __bf16* sent_l = (__bf16*)(ws + O_SENT_L);
  __bf16* a_h    = (__bf16*)(ws + O_A_H);
  __bf16* a_l    = (__bf16*)(ws + O_A_L);
  __bf16* wt_h   = (__bf16*)(ws + O_WT_H);
  __bf16* wt_l   = (__bf16*)(ws + O_WT_L);
  __bf16* ev_h   = (__bf16*)(ws + O_EV_H);
  __bf16* ev_l   = (__bf16*)(ws + O_EV_L);
  __bf16* evt_h  = (__bf16*)(ws + O_EVT_H);
  __bf16* evt_l  = (__bf16*)(ws + O_EVT_L);
  __bf16* lg_h   = (__bf16*)(ws + O_LOG_H);
  __bf16* lg_l   = (__bf16*)(ws + O_LOG_L);

  // stage 1: gathers, norms, pre-swizzled conversions
  k_gather_sent <<<BB * SS, 256, 0, stream>>>(emb, idx_sent, sent, sent_h, sent_l);
  k_gather_event<<<BB * EE, 256, 0, stream>>>(emb, idx_evt, ev_h, ev_l, evt_h, evt_l, ne);
  k_map_num     <<<1, 32, 0, stream>>>(idx_evt, mapn);
  k_convert_w   <<<(DD * DD) / 256, 256, 0, stream>>>(W, wt_h, wt_l);

  // stage 2: a = sent @ W (bf16x2 WMMA, fully coalesced fragment streams)
  k_gemm1<<<dim3(DD / 128, (BB * SS) / 128), 256, 0, stream>>>(
      sent_h, sent_l, wt_h, wt_l, a_h, a_l);
  k_row_norm<<<BB * SS, 256, 0, stream>>>(a_h, a_l, na);

  // stage 3: logits = cos-sim numerator / (na*ne)
  k_gemm2<<<dim3(SS / 128, BB), 256, 0, stream>>>(a_h, a_l, ev_h, ev_l, na, ne, lg_h, lg_l);

  // stage 4: s2e = logits @ event * map_num
  k_gemm3<<<dim3(DD / 128, SS / 128, BB), 256, 0, stream>>>(lg_h, lg_l, evt_h, evt_l, mapn, s2e);

  // stage 5: final linear
  k_final<<<BB * SS, 256, 0, stream>>>(sent, s2e, lin_w, lin_b, pos_tag, out);
}
